// CausalSelfAttention_31705448579574
// MI455X (gfx1250) — compile-verified, matrix operand order checked by reading, not run on hardware
//
#include <hip/hip_runtime.h>
#include <hip/hip_bf16.h>
#include <math.h>

// ---------------------------------------------------------------------------
// CausalSelfAttention for MI455X (gfx1250, wave32, WMMA)
// B=4, T=2048, C=1024, H=16, D=64
// bf16 WMMA (v_wmma_f32_16x16x32_bf16) for all three matmul stages, f32
// softmax. GEMM mainloop uses double-buffered async global->LDS copies
// (GLOBAL_LOAD_ASYNC_TO_LDS_B128 / ASYNCcnt) when the toolchain exposes the
// builtins, falling back to synchronous staging otherwise.
// ---------------------------------------------------------------------------

typedef __bf16 bf16;
typedef __attribute__((ext_vector_type(16))) __bf16 bf16x16;
typedef __attribute__((ext_vector_type(8)))  __bf16 bf16x8;
typedef __attribute__((ext_vector_type(8)))  float  f32x8;
typedef __attribute__((ext_vector_type(4)))  int    i32x4;

#define T_SEQ 2048
#define NHEAD 16
#define HDIM  64
#define CDIM  1024
#define BDIM  4
#define MTOT  (BDIM * T_SEQ)      // 8192 tokens
#define NQKV  (3 * CDIM)          // 3072
#define TILE_ELEMS (128 * 32)     // one LDS staging buffer (bf16 elements)

#if defined(__has_builtin)
#if __has_builtin(__builtin_amdgcn_global_load_async_to_lds_b128)
#define HAVE_ASYNC_COPY 1
#endif
#if __has_builtin(__builtin_amdgcn_s_wait_asynccnt)
#define HAVE_WAIT_ASYNC 1
#endif
#endif

// D = A(16x32) * B(32x16) + C, bf16 in / f32 out
__device__ __forceinline__ f32x8 wmma_bf16(bf16x16 a, bf16x16 b, f32x8 c) {
  return __builtin_amdgcn_wmma_f32_16x16x32_bf16(false, a, false, b, (short)0, c,
                                                 false, false);
}

union frag_u { bf16x16 v; bf16x8 h[2]; };

// Load a 16x32 (row x K) bf16 fragment, row-major with leading dim `ld`.
// ISA 16-bit A/B layout: lane holds row (lane&15); K chunks of 8 at
// base = (lane>>4)*8 and base+16.  Two 16-byte loads per lane.
__device__ __forceinline__ bf16x16 load_frag(const bf16* __restrict__ base, int ld) {
  const int lane = threadIdx.x & 31;
  const int r    = lane & 15;
  const int kb   = (lane >> 4) << 3;   // 0 or 8
  const bf16* p  = base + r * ld + kb;
  frag_u u;
  u.h[0] = *(const bf16x8*)(p);        // K = kb .. kb+7
  u.h[1] = *(const bf16x8*)(p + 16);   // K = kb+16 .. kb+23
  return u.v;
}

// ---- async global->LDS copy of one 16-byte chunk per lane ------------------
// Builtin signature (probe-verified via diagnostics): (int4 AS1*, int4 AS3*,
// imm offset, imm cpol).
__device__ __forceinline__ void async_copy_b128(const bf16* g, bf16* l) {
#if defined(HAVE_ASYNC_COPY)
  __builtin_amdgcn_global_load_async_to_lds_b128(
      (__attribute__((address_space(1))) i32x4*)(void*)(const void*)g,
      (__attribute__((address_space(3))) i32x4*)(void*)l, 0, 0);
#else
  *(bf16x8*)l = *(const bf16x8*)g;
#endif
}

__device__ __forceinline__ void wait_async_le4() {
#if defined(HAVE_WAIT_ASYNC)
  __builtin_amdgcn_s_wait_asynccnt(4);
#else
  asm volatile("s_wait_asynccnt 0x4" ::: "memory");
#endif
}

__device__ __forceinline__ void wait_async_le0() {
#if defined(HAVE_WAIT_ASYNC)
  __builtin_amdgcn_s_wait_asynccnt(0);
#else
  asm volatile("s_wait_asynccnt 0x0" ::: "memory");
#endif
}

// ---------------------------------------------------------------------------
// fp32 -> bf16 conversion kernels
// ---------------------------------------------------------------------------
__global__ __launch_bounds__(256) void cvt_bf16_kernel(const float* __restrict__ src,
                                                       bf16* __restrict__ dst, int n) {
  int i = blockIdx.x * 256 + threadIdx.x;
  if (i < n) dst[i] = (bf16)src[i];
}

// src[rows][cols] (fp32) -> dst[cols][rows] (bf16): store W^T so WMMA B
// fragments load contiguous K.
__global__ __launch_bounds__(256) void cvt_T_kernel(const float* __restrict__ src,
                                                    bf16* __restrict__ dst,
                                                    int rows, int cols) {
  int i = blockIdx.x * 256 + threadIdx.x;
  if (i < rows * cols) {
    int r = i / cols, c = i % cols;
    dst[(size_t)c * rows + r] = (bf16)src[i];
  }
}

// ---------------------------------------------------------------------------
// Shared GEMM mainloop: C[128x128] tile, 256 threads = 8 waves.
// Wave grid 4(M) x 2(N); each wave computes 32x64 = 2x4 WMMA accumulators.
// A is [M][K] row-major bf16, B is W^T = [N][K] row-major bf16.
// Double-buffered: tile i+1 streams into the alternate LDS buffer via async
// copies while WMMA consumes tile i.  Per wave: 4 async ops per tile, so
// s_wait_asynccnt <=4 drains exactly the tile about to be consumed.
// ---------------------------------------------------------------------------
__device__ __forceinline__ void stage_tile(const bf16* ga, const bf16* gb,
                                           bf16* As, bf16* Bs,
                                           int lrow, int lcol) {
  async_copy_b128(ga,     &As[lrow * 32 + lcol]);
  async_copy_b128(ga + 8, &As[lrow * 32 + lcol + 8]);
  async_copy_b128(gb,     &Bs[lrow * 32 + lcol]);
  async_copy_b128(gb + 8, &Bs[lrow * 32 + lcol + 8]);
}

__device__ __forceinline__ void gemm_mainloop(const bf16* __restrict__ A,
                                              const bf16* __restrict__ B,
                                              int K, int m0, int n0,
                                              bf16* As, bf16* Bs,  // [2][TILE_ELEMS]
                                              f32x8 acc[2][4]) {
  const int tid   = threadIdx.x;
  const int wave  = tid >> 5;
  const int waveM = wave & 3;
  const int waveN = wave >> 2;
  const int lrow  = tid >> 1;           // 0..127
  const int lcol  = (tid & 1) << 4;     // 0 or 16

  const bf16* ga = A + (size_t)(m0 + lrow) * K + lcol;
  const bf16* gb = B + (size_t)(n0 + lrow) * K + lcol;

  stage_tile(ga, gb, As, Bs, lrow, lcol);        // tile 0 -> buffer 0
  int cur = 0;

  for (int kk = 0; kk < K; kk += 32) {
    const bool has_next = (kk + 32) < K;
    if (has_next) {
      // stream tile i+1 into the other buffer while tile i computes
      stage_tile(ga + kk + 32, gb + kk + 32,
                 As + (cur ^ 1) * TILE_ELEMS, Bs + (cur ^ 1) * TILE_ELEMS,
                 lrow, lcol);
      __builtin_prefetch(ga + kk + 64, 0, 1);    // global_prefetch_b8 (L2)
      __builtin_prefetch(gb + kk + 64, 0, 1);
      wait_async_le4();                          // drain tile i only
    } else {
      wait_async_le0();
    }
    __syncthreads();

    const bf16* as = As + cur * TILE_ELEMS;
    const bf16* bs = Bs + cur * TILE_ELEMS;
    bf16x16 af0 = load_frag(as + (waveM * 32 +  0) * 32, 32);
    bf16x16 af1 = load_frag(as + (waveM * 32 + 16) * 32, 32);
    bf16x16 bf0 = load_frag(bs + (waveN * 64 +  0) * 32, 32);
    bf16x16 bf1 = load_frag(bs + (waveN * 64 + 16) * 32, 32);
    bf16x16 bf2 = load_frag(bs + (waveN * 64 + 32) * 32, 32);
    bf16x16 bf3 = load_frag(bs + (waveN * 64 + 48) * 32, 32);

    acc[0][0] = wmma_bf16(af0, bf0, acc[0][0]);
    acc[0][1] = wmma_bf16(af0, bf1, acc[0][1]);
    acc[0][2] = wmma_bf16(af0, bf2, acc[0][2]);
    acc[0][3] = wmma_bf16(af0, bf3, acc[0][3]);
    acc[1][0] = wmma_bf16(af1, bf0, acc[1][0]);
    acc[1][1] = wmma_bf16(af1, bf1, acc[1][1]);
    acc[1][2] = wmma_bf16(af1, bf2, acc[1][2]);
    acc[1][3] = wmma_bf16(af1, bf3, acc[1][3]);
    __syncthreads();   // all waves done reading buffer `cur` before it refills
    cur ^= 1;
  }
}

// ---------------------------------------------------------------------------
// QKV GEMM: qkv[m, n] = sum_k x[m,k] * w_qkv[k,n];  n in [0,3072).
// Epilogue scatters into q[b,h,t,d], k[b,h,t,d], v^T[b,h,d,t] (bf16).
// ---------------------------------------------------------------------------
__global__ __launch_bounds__(256) void qkv_gemm_kernel(const bf16* __restrict__ xb,
                                                       const bf16* __restrict__ wT,
                                                       bf16* __restrict__ qb,
                                                       bf16* __restrict__ kb,
                                                       bf16* __restrict__ vt) {
  __shared__ bf16 As[2 * TILE_ELEMS];
  __shared__ bf16 Bs[2 * TILE_ELEMS];
  f32x8 acc[2][4] = {};
  const int m0 = blockIdx.y * 128;
  const int n0 = blockIdx.x * 128;
  gemm_mainloop(xb, wT, CDIM, m0, n0, As, Bs, acc);

  const int tid   = threadIdx.x;
  const int lane  = tid & 31;
  const int wave  = tid >> 5;
  const int waveM = wave & 3;
  const int waveN = wave >> 2;
  const int colL  = lane & 15;
  const int rbase = (lane >> 4) << 3;

  for (int i = 0; i < 2; ++i)
    for (int j = 0; j < 4; ++j)
      for (int r = 0; r < 8; ++r) {
        int m = m0 + waveM * 32 + i * 16 + rbase + r;
        int n = n0 + waveN * 64 + j * 16 + colL;
        bf16 v = (bf16)acc[i][j][r];
        int part = n >> 10;          // 0=q 1=k 2=v
        int hn   = n & 1023;
        int h    = hn >> 6;
        int d    = hn & 63;
        int bb   = m >> 11;          // T=2048
        int t    = m & 2047;
        size_t bh = (size_t)(bb * NHEAD + h);
        if (part == 0)      qb[(bh * T_SEQ + t) * HDIM + d] = v;
        else if (part == 1) kb[(bh * T_SEQ + t) * HDIM + d] = v;
        else                vt[(bh * HDIM + d) * T_SEQ + t] = v;
      }
}

// ---------------------------------------------------------------------------
// RoPE on q and k in place. One thread per (b,h,t,j) pair with j in [0,32).
// ---------------------------------------------------------------------------
__global__ __launch_bounds__(256) void rope_kernel(bf16* __restrict__ q,
                                                   bf16* __restrict__ k) {
  int idx = blockIdx.x * 256 + threadIdx.x;
  int j   = idx & 31;
  int t   = (idx >> 5) & (T_SEQ - 1);
  int bh  = idx >> 16;
  if (bh >= BDIM * NHEAD) return;
  float inv = __powf(10000.0f, -(float)(2 * j) * (1.0f / 64.0f));
  float ang = (float)t * inv;
  float s, c;
  __sincosf(ang, &s, &c);
  size_t base = ((size_t)bh * T_SEQ + t) * HDIM;
  float q1 = (float)q[base + j], q2 = (float)q[base + j + 32];
  q[base + j]      = (bf16)(q1 * c - q2 * s);
  q[base + j + 32] = (bf16)(q2 * c + q1 * s);
  float k1 = (float)k[base + j], k2 = (float)k[base + j + 32];
  k[base + j]      = (bf16)(k1 * c - k2 * s);
  k[base + j + 32] = (bf16)(k2 * c + k1 * s);
}

// ---------------------------------------------------------------------------
// Flash attention: grid (T/64, B*H), 128 threads = 4 waves, each wave owns
// 16 query rows. Online softmax in f32, WMMA for Q.K^T and P.V.
// No block barriers in the mainloop (waves have divergent causal trip counts);
// P goes through a per-wave LDS slab guarded by s_wait_dscnt.
// ---------------------------------------------------------------------------
__global__ __launch_bounds__(128) void flash_kernel(const bf16* __restrict__ qb,
                                                    const bf16* __restrict__ kb,
                                                    const bf16* __restrict__ vt,
                                                    bf16* __restrict__ yb) {
  __shared__ bf16 pls[4][16 * 32];
  const int tid  = threadIdx.x;
  const int lane = tid & 31;
  const int wave = tid >> 5;
  const int bh   = blockIdx.y;
  const int q0   = blockIdx.x * 64 + wave * 16;
  const bf16* qh = qb + (size_t)bh * T_SEQ * HDIM;
  const bf16* kh = kb + (size_t)bh * T_SEQ * HDIM;
  const bf16* vh = vt + (size_t)bh * HDIM * T_SEQ;
  const int colL  = lane & 15;
  const int rbase = (lane >> 4) << 3;
  const float scale = 0.125f;   // 1/sqrt(64)

  bf16x16 aq0 = load_frag(qh + (size_t)q0 * HDIM, HDIM);        // d 0..31
  bf16x16 aq1 = load_frag(qh + (size_t)q0 * HDIM + 32, HDIM);   // d 32..63

  f32x8 o0 = {}, o1 = {}, o2 = {}, o3 = {};
  float mrow[8], lrow[8];
  for (int r = 0; r < 8; ++r) { mrow[r] = -1e30f; lrow[r] = 0.0f; }

  bf16* myp = pls[wave];

  for (int kb0 = 0; kb0 < q0 + 16; kb0 += 32) {
    // ---- S = Q * K^T  (16 x 32 scores = two 16x16 tiles) ----
    f32x8 s0 = {}, s1 = {};
    {
      bf16x16 bk00 = load_frag(kh + (size_t)(kb0 +  0) * HDIM +  0, HDIM);
      bf16x16 bk10 = load_frag(kh + (size_t)(kb0 + 16) * HDIM +  0, HDIM);
      s0 = wmma_bf16(aq0, bk00, s0);
      s1 = wmma_bf16(aq0, bk10, s1);
      bf16x16 bk01 = load_frag(kh + (size_t)(kb0 +  0) * HDIM + 32, HDIM);
      bf16x16 bk11 = load_frag(kh + (size_t)(kb0 + 16) * HDIM + 32, HDIM);
      s0 = wmma_bf16(aq1, bk01, s0);
      s1 = wmma_bf16(aq1, bk11, s1);
    }

    // ---- causal mask + online softmax update (per row) ----
    for (int r = 0; r < 8; ++r) {
      int qi = q0 + rbase + r;
      float a  = (kb0 + colL      <= qi) ? s0[r] * scale : -1e30f;
      float bb = (kb0 + colL + 16 <= qi) ? s1[r] * scale : -1e30f;
      float rmax = fmaxf(a, bb);
      rmax = fmaxf(rmax, __shfl_xor(rmax, 1));
      rmax = fmaxf(rmax, __shfl_xor(rmax, 2));
      rmax = fmaxf(rmax, __shfl_xor(rmax, 4));
      rmax = fmaxf(rmax, __shfl_xor(rmax, 8));
      float mnew  = fmaxf(mrow[r], rmax);
      float alpha = __expf(mrow[r] - mnew);
      float p0 = __expf(a - mnew);
      float p1 = __expf(bb - mnew);
      float rs = p0 + p1;
      rs += __shfl_xor(rs, 1);
      rs += __shfl_xor(rs, 2);
      rs += __shfl_xor(rs, 4);
      rs += __shfl_xor(rs, 8);
      lrow[r] = lrow[r] * alpha + rs;
      mrow[r] = mnew;
      o0[r] = o0[r] * alpha;
      o1[r] = o1[r] * alpha;
      o2[r] = o2[r] * alpha;
      o3[r] = o3[r] * alpha;
      // stash probabilities (row-major 16x32 bf16) for the P.V fragment
      myp[(rbase + r) * 32 + colL]      = (bf16)p0;
      myp[(rbase + r) * 32 + colL + 16] = (bf16)p1;
    }
    // wave-private LDS slab: DS ops are in-order per wave; drain before reload
    asm volatile("s_wait_dscnt 0x0" ::: "memory");

    // ---- O += P * V  (P 16x32, V^T rows are contiguous in t) ----
    bf16x16 ap = load_frag(myp, 32);
    o0 = wmma_bf16(ap, load_frag(vh + (size_t)( 0) * T_SEQ + kb0, T_SEQ), o0);
    o1 = wmma_bf16(ap, load_frag(vh + (size_t)(16) * T_SEQ + kb0, T_SEQ), o1);
    o2 = wmma_bf16(ap, load_frag(vh + (size_t)(32) * T_SEQ + kb0, T_SEQ), o2);
    o3 = wmma_bf16(ap, load_frag(vh + (size_t)(48) * T_SEQ + kb0, T_SEQ), o3);
  }

  // ---- epilogue: y[b, t, h*64 + d] = O / l ----
  const int bb = bh >> 4;
  const int h  = bh & 15;
  for (int r = 0; r < 8; ++r) {
    float invl = 1.0f / lrow[r];
    int t = q0 + rbase + r;
    size_t base = ((size_t)bb * T_SEQ + t) * CDIM + h * HDIM + colL;
    yb[base +  0] = (bf16)(o0[r] * invl);
    yb[base + 16] = (bf16)(o1[r] * invl);
    yb[base + 32] = (bf16)(o2[r] * invl);
    yb[base + 48] = (bf16)(o3[r] * invl);
  }
}

// ---------------------------------------------------------------------------
// Output projection: out[m, n] = sum_k y[m,k] * w_proj[k,n], f32 output.
// ---------------------------------------------------------------------------
__global__ __launch_bounds__(256) void proj_gemm_kernel(const bf16* __restrict__ yb,
                                                        const bf16* __restrict__ wT,
                                                        float* __restrict__ out) {
  __shared__ bf16 As[2 * TILE_ELEMS];
  __shared__ bf16 Bs[2 * TILE_ELEMS];
  f32x8 acc[2][4] = {};
  const int m0 = blockIdx.y * 128;
  const int n0 = blockIdx.x * 128;
  gemm_mainloop(yb, wT, CDIM, m0, n0, As, Bs, acc);

  const int tid   = threadIdx.x;
  const int lane  = tid & 31;
  const int wave  = tid >> 5;
  const int waveM = wave & 3;
  const int waveN = wave >> 2;
  const int colL  = lane & 15;
  const int rbase = (lane >> 4) << 3;

  for (int i = 0; i < 2; ++i)
    for (int j = 0; j < 4; ++j)
      for (int r = 0; r < 8; ++r) {
        int m = m0 + waveM * 32 + i * 16 + rbase + r;
        int n = n0 + waveN * 64 + j * 16 + colL;
        out[(size_t)m * CDIM + n] = acc[i][j][r];
      }
}

// ---------------------------------------------------------------------------
// Host-side launch
// ---------------------------------------------------------------------------
extern "C" void kernel_launch(void* const* d_in, const int* in_sizes, int n_in,
                              void* d_out, int out_size, void* d_ws, size_t ws_size,
                              hipStream_t stream) {
  const float* x  = (const float*)d_in[0];   // [8192,1024]
  const float* wq = (const float*)d_in[1];   // [1024,3072]
  const float* wp = (const float*)d_in[2];   // [1024,1024]
  float* out = (float*)d_out;                // [8192,1024] f32
  char* ws = (char*)d_ws;
  const size_t MB = (size_t)1 << 20;

  bf16* xb     = (bf16*)(ws + 0 * MB);   // 16 MiB  x in bf16
  bf16* wqkvT  = (bf16*)(ws + 16 * MB);  //  6 MiB  w_qkv^T bf16
  bf16* wprojT = (bf16*)(ws + 22 * MB);  //  2 MiB  w_proj^T bf16
  bf16* qb     = (bf16*)(ws + 24 * MB);  // 16 MiB  q[b,h,t,d]
  bf16* kbuf   = (bf16*)(ws + 40 * MB);  // 16 MiB  k[b,h,t,d]
  bf16* vt     = (bf16*)(ws + 56 * MB);  // 16 MiB  v^T[b,h,d,t]
  bf16* yb     = (bf16*)(ws + 72 * MB);  // 16 MiB  attn out [b*t, 1024]

  cvt_bf16_kernel<<<(MTOT * CDIM) / 256, 256, 0, stream>>>(x, xb, MTOT * CDIM);
  cvt_T_kernel<<<(CDIM * NQKV) / 256, 256, 0, stream>>>(wq, wqkvT, CDIM, NQKV);
  cvt_T_kernel<<<(CDIM * CDIM) / 256, 256, 0, stream>>>(wp, wprojT, CDIM, CDIM);

  qkv_gemm_kernel<<<dim3(NQKV / 128, MTOT / 128), 256, 0, stream>>>(xb, wqkvT, qb,
                                                                    kbuf, vt);

  rope_kernel<<<(BDIM * NHEAD * T_SEQ * 32) / 256, 256, 0, stream>>>(qb, kbuf);

  flash_kernel<<<dim3(T_SEQ / 64, BDIM * NHEAD), 128, 0, stream>>>(qb, kbuf, vt, yb);

  proj_gemm_kernel<<<dim3(CDIM / 128, MTOT / 128), 256, 0, stream>>>(yb, wprojT, out);
}
